// FlashGQAAttention_50903952392560
// MI455X (gfx1250) — matrix operand order repaired and checked by reading, not compiled, not run on hardware
//
#include <hip/hip_runtime.h>
#include <hip/hip_bf16.h>
#include <stdint.h>
#include <stddef.h>

typedef __attribute__((ext_vector_type(16))) _Float16 v16h;
typedef __attribute__((ext_vector_type(8)))  float    v8f;

#define SEQ 2048
#define HID 1024
#define NH  16
#define NKV 4
#define HD  64
#define BS  4

struct __align__(16) U4 { unsigned x, y, z, w; };

union FragA { unsigned u[8]; v16h v; };     // A-matrix fragment (b32 gathers)
union FragB { U4 q[2]; v16h v; };           // B-matrix fragment (2x b128)
union H2    { unsigned u; _Float16 h[2]; };
union X8    { U4 q; _Float16 h[8]; };

// A-fragment k order used here (contiguous k = 16*(lane/16) + 2p, same as B): element pair p (halves 2p,2p+1) -> starting K
__device__ __forceinline__ int kpairA(int p, int lane) {
    return (p << 1) | (lane & 16);
}

__device__ __forceinline__ v8f wmma_f16(v16h a, v16h b, v8f c) {
    return __builtin_amdgcn_wmma_f32_16x16x32_f16(false, a, false, b,
                                                  (short)0, c, false, false);
}

// ---------------------------------------------------------------- converts
__global__ void convert_f32_to_f16(const float* __restrict__ in,
                                   _Float16* __restrict__ out, int n) {
    int i = blockIdx.x * blockDim.x + threadIdx.x;
    if (i < n) out[i] = (_Float16)in[i];
}

__global__ void pack_qkv_weights(const float* __restrict__ Wq,
                                 const float* __restrict__ Wk,
                                 const float* __restrict__ Wv,
                                 _Float16* __restrict__ out) {
    int i = blockIdx.x * blockDim.x + threadIdx.x;          // over 1024*1536
    if (i >= HID * 1536) return;
    int n = i % 1536, k = i / 1536;
    float v;
    if (n < 1024)      v = Wq[k * 1024 + n];
    else if (n < 1280) v = Wk[k * 256 + (n - 1024)];
    else               v = Wv[k * 256 + (n - 1280)];
    out[i] = (_Float16)v;
}

// ---------------------------------------------------------------- GEMM
// C[M,N] = A[M,K] * B[K,N]; block tile 128x64, BK=32, 8 waves of 32x32.
// mode 0: outf[m*N+n] = acc + bias0[n]           (f32 output)
// mode 1: scatter f16 into q/k/v [b][h][s][64]   (+ per-segment bias)
__global__ void __launch_bounds__(256) gemm_wmma(
    const _Float16* __restrict__ A, const _Float16* __restrict__ B,
    int K, int N, int mode,
    const float* __restrict__ bias0, const float* __restrict__ bias1,
    const float* __restrict__ bias2,
    float* __restrict__ outf,
    _Float16* __restrict__ qh, _Float16* __restrict__ kh,
    _Float16* __restrict__ vh)
{
    __shared__ _Float16 Ash[128 * 40];   // [row][k], 80B row stride (16B-aligned)
    __shared__ _Float16 Bt[64 * 40];     // transposed [n][k], 80B row stride

    const int tid  = threadIdx.x;
    const int lane = tid & 31;
    const int w    = tid >> 5;
    const int ln   = lane & 15;
    const int hi16 = lane & 16;
    const int waveM = w >> 1, waveN = w & 1;
    const int m0 = blockIdx.y * 128, n0 = blockIdx.x * 64;

    v8f acc[2][2];
#pragma unroll
    for (int i = 0; i < 2; ++i)
#pragma unroll
        for (int j = 0; j < 2; ++j)
#pragma unroll
            for (int e = 0; e < 8; ++e) acc[i][j][e] = 0.0f;

    for (int k0 = 0; k0 < K; k0 += 32) {
        __syncthreads();
        // A tile: 128x32 halves = 512 x b128, two per thread
#pragma unroll
        for (int r = 0; r < 2; ++r) {
            int idx = (tid + r * 256) * 8;
            int row = idx >> 5, kk = idx & 31;
            U4 d = *(const U4*)(A + (size_t)(m0 + row) * K + k0 + kk);
            unsigned* dst = (unsigned*)&Ash[row * 40 + kk];
            dst[0] = d.x; dst[1] = d.y; dst[2] = d.z; dst[3] = d.w;
        }
        // B tile: 32x64 halves, store transposed for contiguous fragments
        {
            int idx = tid * 8;
            int kk = idx >> 6, nn = idx & 63;
            X8 d; d.q = *(const U4*)(B + (size_t)(k0 + kk) * N + n0 + nn);
#pragma unroll
            for (int t = 0; t < 8; ++t) Bt[(nn + t) * 40 + kk] = d.h[t];
        }
        __syncthreads();

        FragA af[2];
        FragB bf[2];
#pragma unroll
        for (int i = 0; i < 2; ++i) {
            int row = waveM * 32 + i * 16 + ln;
#pragma unroll
            for (int p = 0; p < 8; ++p)
                af[i].u[p] = *(const unsigned*)&Ash[row * 40 + kpairA(p, lane)];
        }
#pragma unroll
        for (int j = 0; j < 2; ++j) {
            int col = waveN * 32 + j * 16 + ln;
            const U4* src = (const U4*)&Bt[col * 40 + hi16];
            bf[j].q[0] = src[0]; bf[j].q[1] = src[1];
        }
#pragma unroll
        for (int i = 0; i < 2; ++i)
#pragma unroll
            for (int j = 0; j < 2; ++j)
                acc[i][j] = wmma_f16(af[i].v, bf[j].v, acc[i][j]);
    }

    // epilogue: C/D layout row = v + 8*(lane>=16), col = lane&15
#pragma unroll
    for (int i = 0; i < 2; ++i) {
#pragma unroll
        for (int j = 0; j < 2; ++j) {
#pragma unroll
            for (int v = 0; v < 8; ++v) {
                int m = m0 + waveM * 32 + i * 16 + v + (hi16 >> 1);
                int n = n0 + waveN * 32 + j * 16 + ln;
                float val = acc[i][j][v];
                if (mode == 0) {
                    outf[(size_t)m * N + n] = val + bias0[n];
                } else {
                    int b = m >> 11, s = m & 2047;
                    if (n < 1024) {
                        int hh = n >> 6, dd = n & 63;
                        qh[(((size_t)(b * NH + hh)) * SEQ + s) * HD + dd] =
                            (_Float16)(val + bias0[n]);
                    } else if (n < 1280) {
                        int nk = n - 1024, hh = nk >> 6, dd = nk & 63;
                        kh[(((size_t)(b * NKV + hh)) * SEQ + s) * HD + dd] =
                            (_Float16)(val + bias1[nk]);
                    } else {
                        int nv = n - 1280, hh = nv >> 6, dd = nv & 63;
                        vh[(((size_t)(b * NKV + hh)) * SEQ + s) * HD + dd] =
                            (_Float16)(val + bias2[nv]);
                    }
                }
            }
        }
    }
}

// ---------------------------------------------------------------- RoPE
__global__ void rope_kernel(_Float16* __restrict__ buf,
                            const int* __restrict__ pos_ids,
                            int heads, int ntot) {
    int idx = blockIdx.x * blockDim.x + threadIdx.x;
    if (idx >= ntot) return;
    int i  = idx & 31;
    int s  = (idx >> 5) & (SEQ - 1);
    int bh = idx >> 16;                      // 5 + 11 bits
    int b  = bh / heads;
    float p = (float)pos_ids[b * SEQ + s];
    // inv_freq[i] = 10000^(-2i/64) = exp(-i * ln(10000) / 32)
    float ang = p * __expf((float)i * (-9.210340371976184f / 32.0f));
    float sn, cs;
    __sincosf(ang, &sn, &cs);
    size_t o0 = (size_t)bh * SEQ * HD + (size_t)s * HD + i;
    float v0 = (float)buf[o0];
    float v1 = (float)buf[o0 + 32];
    buf[o0]      = (_Float16)(v0 * cs - v1 * sn);
    buf[o0 + 32] = (_Float16)(v1 * cs + v0 * sn);
}

// ---------------------------------------------------------------- attention
// One wave = 16 q rows x full D=64. Block = 8 waves share K/V LDS tiles.
__global__ void __launch_bounds__(256) attn_kernel(
    const _Float16* __restrict__ qh, const _Float16* __restrict__ kh,
    const _Float16* __restrict__ vh, _Float16* __restrict__ attnh)
{
    __shared__ _Float16 Ksh[32 * 72];        // [kv][d], 144B row stride
    __shared__ _Float16 Vt[64 * 40];         // transposed [d][kv], 80B stride
    __shared__ _Float16 Psh[8 * 16 * 34];    // per-wave P scratch

    const int tid  = threadIdx.x;
    const int lane = tid & 31;
    const int w    = tid >> 5;
    const int ln   = lane & 15;
    const int hi16 = lane & 16;
    const int hi8  = hi16 >> 1;

    const int bh = blockIdx.y;
    const int b = bh / NH, h = bh % NH;
    const int hk = h / (NH / NKV);
    const int q0 = blockIdx.x * 128 + w * 16;

    const _Float16* qbase = qh + (size_t)bh * SEQ * HD;
    const _Float16* kbase = kh + (size_t)(b * NKV + hk) * SEQ * HD;
    const _Float16* vbase = vh + (size_t)(b * NKV + hk) * SEQ * HD;

    // preload Q fragments, pre-scaled by 1/sqrt(D)
    FragA qf[2];
    {
        const _Float16* qrow = qbase + (size_t)(q0 + ln) * HD;
#pragma unroll
        for (int ks = 0; ks < 2; ++ks) {
#pragma unroll
            for (int p = 0; p < 8; ++p) {
                int d = ks * 32 + kpairA(p, lane);
                H2 t; t.u = *(const unsigned*)(qrow + d);
                t.h[0] = (_Float16)((float)t.h[0] * 0.125f);
                t.h[1] = (_Float16)((float)t.h[1] * 0.125f);
                qf[ks].u[p] = t.u;
            }
        }
    }

    float rm[8], rl[8];
    v8f o[4];
#pragma unroll
    for (int v = 0; v < 8; ++v) { rm[v] = -1e30f; rl[v] = 0.0f; }
#pragma unroll
    for (int nt = 0; nt < 4; ++nt)
#pragma unroll
        for (int v = 0; v < 8; ++v) o[nt][v] = 0.0f;

    for (int c = 0; c < SEQ / 32; ++c) {
        const int kv0 = c * 32;
        __syncthreads();
        {   // cooperative stage of K (row-major) and V (transposed)
            int idx = tid * 8;
            int kvl = idx >> 6, dd = idx & 63;
            U4 k4 = *(const U4*)(kbase + (size_t)(kv0 + kvl) * HD + dd);
            unsigned* kd = (unsigned*)&Ksh[kvl * 72 + dd];
            kd[0] = k4.x; kd[1] = k4.y; kd[2] = k4.z; kd[3] = k4.w;
            X8 v8; v8.q = *(const U4*)(vbase + (size_t)(kv0 + kvl) * HD + dd);
#pragma unroll
            for (int t = 0; t < 8; ++t) Vt[(dd + t) * 40 + kvl] = v8.h[t];
        }
        if (c + 1 < SEQ / 32) {   // pull next K/V chunk toward L2 (global_prefetch_b8)
            __builtin_prefetch(kbase + (size_t)(kv0 + 32) * HD + tid * 8, 0, 1);
            __builtin_prefetch(vbase + (size_t)(kv0 + 32) * HD + tid * 8, 0, 1);
        }
        __syncthreads();

        // S = Q * K^T  (two 16x16 score tiles over the 32 kv chunk)
        v8f s0, s1;
#pragma unroll
        for (int e = 0; e < 8; ++e) { s0[e] = 0.0f; s1[e] = 0.0f; }
#pragma unroll
        for (int ks = 0; ks < 2; ++ks) {
            FragB kf0, kf1;
            const U4* r0 = (const U4*)&Ksh[ln * 72 + ks * 32 + hi16];
            kf0.q[0] = r0[0]; kf0.q[1] = r0[1];
            const U4* r1 = (const U4*)&Ksh[(16 + ln) * 72 + ks * 32 + hi16];
            kf1.q[0] = r1[0]; kf1.q[1] = r1[1];
            s0 = wmma_f16(qf[ks].v, kf0.v, s0);
            s1 = wmma_f16(qf[ks].v, kf1.v, s1);
        }

        // online softmax: element v holds row (v + hi8), col = ln (+16 for s1)
#pragma unroll
        for (int v = 0; v < 8; ++v) {
            float mx = fmaxf(s0[v], s1[v]);
            mx = fmaxf(mx, __shfl_xor(mx, 1));
            mx = fmaxf(mx, __shfl_xor(mx, 2));
            mx = fmaxf(mx, __shfl_xor(mx, 4));
            mx = fmaxf(mx, __shfl_xor(mx, 8));
            float mnew = fmaxf(rm[v], mx);
            float corr = __expf(rm[v] - mnew);
            float p0 = __expf(s0[v] - mnew);
            float p1 = __expf(s1[v] - mnew);
            float sum = p0 + p1;
            sum += __shfl_xor(sum, 1);
            sum += __shfl_xor(sum, 2);
            sum += __shfl_xor(sum, 4);
            sum += __shfl_xor(sum, 8);
            rl[v] = rl[v] * corr + sum;
            rm[v] = mnew;
            s0[v] = p0; s1[v] = p1;
#pragma unroll
            for (int nt = 0; nt < 4; ++nt) o[nt][v] *= corr;
        }

        // P: D-layout -> A-fragment layout via per-wave LDS pad
        _Float16* pw = Psh + w * 16 * 34;
#pragma unroll
        for (int v = 0; v < 8; ++v) {
            pw[(v + hi8) * 34 + ln]      = (_Float16)s0[v];
            pw[(v + hi8) * 34 + ln + 16] = (_Float16)s1[v];
        }
        asm volatile("s_wait_dscnt 0" ::: "memory");
        FragA pf;
#pragma unroll
        for (int p = 0; p < 8; ++p)
            pf.u[p] = *(const unsigned*)&pw[ln * 34 + kpairA(p, lane)];

        // O += P * V  (four 16-wide d tiles, K = 32 kv)
#pragma unroll
        for (int nt = 0; nt < 4; ++nt) {
            FragB vf;
            const U4* rv = (const U4*)&Vt[(nt * 16 + ln) * 40 + hi16];
            vf.q[0] = rv[0]; vf.q[1] = rv[1];
            o[nt] = wmma_f16(pf.v, vf.v, o[nt]);
        }
    }

    // normalize and write f16 output [b][s][h*64+d]
#pragma unroll
    for (int nt = 0; nt < 4; ++nt) {
#pragma unroll
        for (int v = 0; v < 8; ++v) {
            int m = q0 + v + hi8;
            int d = nt * 16 + ln;
            attnh[((size_t)(b * SEQ + m)) * (NH * HD) + h * HD + d] =
                (_Float16)(o[nt][v] / rl[v]);
        }
    }
}

// ---------------------------------------------------------------- launch
extern "C" void kernel_launch(void* const* d_in, const int* in_sizes, int n_in,
                              void* d_out, int out_size, void* d_ws, size_t ws_size,
                              hipStream_t stream) {
    (void)in_sizes; (void)n_in; (void)out_size; (void)ws_size;
    const float* x   = (const float*)d_in[0];
    const int*   pos = (const int*)d_in[1];
    const float* Wq  = (const float*)d_in[2];
    const float* bq  = (const float*)d_in[3];
    const float* Wk  = (const float*)d_in[4];
    const float* bk  = (const float*)d_in[5];
    const float* Wv  = (const float*)d_in[6];
    const float* bv  = (const float*)d_in[7];
    const float* Wo  = (const float*)d_in[8];
    const float* bo  = (const float*)d_in[9];

    char* ws = (char*)d_ws;
    size_t off = 0;
    _Float16* xh    = (_Float16*)(ws + off); off += (size_t)BS * SEQ * HID * 2;
    _Float16* wqkvh = (_Float16*)(ws + off); off += (size_t)HID * 1536 * 2;
    _Float16* woh   = (_Float16*)(ws + off); off += (size_t)HID * HID * 2;
    _Float16* qhp   = (_Float16*)(ws + off); off += (size_t)BS * NH * SEQ * HD * 2;
    _Float16* khp   = (_Float16*)(ws + off); off += (size_t)BS * NKV * SEQ * HD * 2;
    _Float16* vhp   = (_Float16*)(ws + off); off += (size_t)BS * NKV * SEQ * HD * 2;
    _Float16* attnh = (_Float16*)(ws + off); off += (size_t)BS * SEQ * HID * 2;

    // 1. precision conversion / packing
    convert_f32_to_f16<<<(BS * SEQ * HID) / 256, 256, 0, stream>>>(
        x, xh, BS * SEQ * HID);
    pack_qkv_weights<<<(HID * 1536) / 256, 256, 0, stream>>>(Wq, Wk, Wv, wqkvh);
    convert_f32_to_f16<<<(HID * HID) / 256, 256, 0, stream>>>(Wo, woh, HID * HID);

    // 2. fused QKV projection (M=8192, K=1024, N=1536) -> q/k/v f16 layouts
    gemm_wmma<<<dim3(1536 / 64, (BS * SEQ) / 128), 256, 0, stream>>>(
        xh, wqkvh, HID, 1536, 1, bq, bk, bv, nullptr, qhp, khp, vhp);

    // 3. RoPE on q and k
    rope_kernel<<<(BS * NH * SEQ * 32) / 256, 256, 0, stream>>>(
        qhp, pos, NH, BS * NH * SEQ * 32);
    rope_kernel<<<(BS * NKV * SEQ * 32) / 256, 256, 0, stream>>>(
        khp, pos, NKV, BS * NKV * SEQ * 32);

    // 4. flash attention (non-causal, online softmax), f32 accumulators
    attn_kernel<<<dim3(SEQ / 128, BS * NH), 256, 0, stream>>>(
        qhp, khp, vhp, attnh);

    // 5. output projection (M=8192, K=1024, N=1024) -> f32 d_out
    gemm_wmma<<<dim3(HID / 64, (BS * SEQ) / 128), 256, 0, stream>>>(
        attnh, woh, HID, HID, 0, bo, nullptr, nullptr,
        (float*)d_out, nullptr, nullptr, nullptr);
}
